// TensorProductConvModel_73426760893079
// MI455X (gfx1250) — matrix operand order repaired and checked by reading, not compiled, not run on hardware
//
#include <hip/hip_runtime.h>
#include <math.h>

typedef __attribute__((ext_vector_type(16))) _Float16 v16h;
typedef __attribute__((ext_vector_type(8)))  float    v8f;
typedef __attribute__((ext_vector_type(4)))  unsigned int u32x4;
typedef __attribute__((ext_vector_type(4)))  int      i32x4;
typedef __attribute__((ext_vector_type(8)))  int      i32x8;

#define NNODES   10000
#define NEDGES   128000
#define RDIM_    32
#define RMAX_    12.0f

#define A0SCALE  0.22360679774997896f   // 1/sqrt(ns+nv)  = 1/sqrt(20)
#define A1SCALE  0.20412414523193154f   // 1/sqrt(ns+2nv) = 1/sqrt(24)
#define INV_S3   0.57735026918962576f
#define INV_S2   0.70710678118654752f
#define SQRT3_   1.73205080756887729f

// per-layer pre-swizzled weight fragments: fc1 (4096 halves) | fc2 (26624 halves)
#define WFRAG_PER_LAYER 30720          // halves  (= 61440 bytes = 7680 x 8B)
#define WFRAG_QWORDS    7680

#if __has_builtin(__builtin_amdgcn_tensor_load_to_lds) && \
    __has_builtin(__builtin_amdgcn_s_wait_tensorcnt)
#define HAVE_TDM 1
#else
#define HAVE_TDM 0
#endif

// ---------------------------------------------------------------- utilities

__device__ inline void atomAddF(float* p, float v) { unsafeAtomicAdd(p, v); }

__device__ inline v8f wmma16(v16h a, v16h b, v8f c) {
  // v_wmma_f32_16x16x32_f16
  return __builtin_amdgcn_wmma_f32_16x16x32_f16(false, a, false, b, (short)0, c,
                                                false, false);
}

// A-fragment (16x32 f16, M x K) from LDS row-major tile with stride 68 halves.
__device__ inline v16h load_afrag(const _Float16* A, int lane, int kstep) {
  const int m = lane & 15;
  const int kbase = (lane < 16) ? 0 : 16;
  v16h a;
#pragma unroll
  for (int j = 0; j < 16; ++j) {
    int koff = kbase + j;
    a[j] = A[m * 68 + 32 * kstep + koff];
  }
  return a;
}

// B-fragment: pre-swizzled so each lane reads 16 contiguous halves (2x b128).
__device__ inline v16h load_bfrag(const _Float16* p, int lane) {
  return *(const v16h*)(p + lane * 16);
}

// ---------------------------------------------------------------- small kernels

__global__ void k_zero(float* __restrict__ p, int n) {
  int i = blockIdx.x * blockDim.x + threadIdx.x;
  if (i < n) p[i] = 0.0f;
}

// Pre-swizzle fc1 (pad K 48->64) and fc2 into per-layer contiguous WMMA
// B-fragment layout (f16):  layer base l*30720 ;
//   fc1 at (s*4+j)*512 + t*16 + e ; fc2 at 4096 + (s*26+j)*512 + t*16 + e
//   k = 32*s + ((t>=16)?16:0) + e ; n = 16*j + (t&15)
__global__ void k_prep(const float* __restrict__ fc1W, const float* __restrict__ fc2W,
                       _Float16* __restrict__ wfrag) {
  int i = blockIdx.x * blockDim.x + threadIdx.x;
  if (i >= 2 * WFRAG_PER_LAYER) return;
  int l = i / WFRAG_PER_LAYER, r = i % WFRAG_PER_LAYER;
  if (r < 4096) {
    int e = r & 15, t = (r >> 4) & 31, j = (r >> 9) & 3, s = (r >> 11) & 1;
    int k = 32 * s + ((t >= 16) ? 16 : 0) + e;
    int n = 16 * j + (t & 15);
    wfrag[i] = (_Float16)((k < 48) ? fc1W[(l * 48 + k) * 64 + n] : 0.0f);
  } else {
    int rr = r - 4096;
    int e = rr & 15, t = (rr >> 4) & 31;
    int rest = rr >> 9;  // s*26 + j
    int j = rest % 26, s = rest / 26;
    int k = 32 * s + ((t >= 16) ? 16 : 0) + e;
    int n = 16 * j + (t & 15);
    wfrag[i] = (_Float16)fc2W[(l * 64 + k) * 416 + n];
  }
}

// h0 = MLP(node_feats @ enc_W + enc_b)
__global__ void k_node_embed(const float* __restrict__ nf,
                             const float* __restrict__ encW, const float* __restrict__ encB,
                             const float* __restrict__ W1, const float* __restrict__ b1,
                             const float* __restrict__ W2, const float* __restrict__ b2,
                             float* __restrict__ h0) {
  int n = blockIdx.x * blockDim.x + threadIdx.x;
  if (n >= NNODES) return;
  float x[64];
#pragma unroll 4
  for (int k = 0; k < 64; ++k) x[k] = nf[n * 64 + k];
  float t0[16], t1[16];
  for (int j = 0; j < 16; ++j) {
    float a = encB[j];
    for (int k = 0; k < 64; ++k) a += x[k] * encW[k * 16 + j];
    t0[j] = a;
  }
  for (int j = 0; j < 16; ++j) {
    float a = b1[j];
    for (int k = 0; k < 16; ++k) a += t0[k] * W1[k * 16 + j];
    t1[j] = fmaxf(a, 0.0f);
  }
  for (int j = 0; j < 16; ++j) {
    float a = b2[j];
    for (int k = 0; k < 16; ++k) a += t1[k] * W2[k * 16 + j];
    h0[n * 16 + j] = a;
  }
}

// edge_attr = MLP(edge_attr_in) + MLP(rbf(d)); y1 = sqrt(3)*unit ; cnt[src]+=1
__global__ void k_edge_prep(const float* __restrict__ pos, const float* __restrict__ eain,
                            const int* __restrict__ ei,
                            const float* __restrict__ eeW1, const float* __restrict__ eeb1,
                            const float* __restrict__ eeW2, const float* __restrict__ eeb2,
                            const float* __restrict__ reW1, const float* __restrict__ reb1,
                            const float* __restrict__ reW2, const float* __restrict__ reb2,
                            float* __restrict__ edge_attr, float* __restrict__ y1g,
                            float* __restrict__ cntf) {
  int e = blockIdx.x * blockDim.x + threadIdx.x;
  if (e >= NEDGES) return;
  int s = ei[e], d = ei[NEDGES + e];
  float vx = pos[s * 3 + 0] - pos[d * 3 + 0];
  float vy = pos[s * 3 + 1] - pos[d * 3 + 1];
  float vz = pos[s * 3 + 2] - pos[d * 3 + 2];
  float dist = sqrtf(vx * vx + vy * vy + vz * vz);
  float inv = 1.0f / fmaxf(dist, 1e-9f);
  y1g[e * 3 + 0] = SQRT3_ * vx * inv;
  y1g[e * 3 + 1] = SQRT3_ * vy * inv;
  y1g[e * 3 + 2] = SQRT3_ * vz * inv;

  const float step = RMAX_ / (RDIM_ - 1);
  const float coeff = -0.5f / (step * step);
  float rbf[RDIM_];
#pragma unroll 4
  for (int i = 0; i < RDIM_; ++i) {
    float t = dist - step * (float)i;
    rbf[i] = __expf(coeff * t * t);
  }
  float xin[16];
#pragma unroll
  for (int k = 0; k < 16; ++k) xin[k] = eain[e * 16 + k];
  float ta[16], attr[16], tb[16];
  for (int j = 0; j < 16; ++j) {
    float a = eeb1[j];
    for (int k = 0; k < 16; ++k) a += xin[k] * eeW1[k * 16 + j];
    ta[j] = fmaxf(a, 0.0f);
  }
  for (int j = 0; j < 16; ++j) {
    float a = eeb2[j];
    for (int k = 0; k < 16; ++k) a += ta[k] * eeW2[k * 16 + j];
    attr[j] = a;
  }
  for (int j = 0; j < 16; ++j) {
    float a = reb1[j];
    for (int k = 0; k < 32; ++k) a += rbf[k] * reW1[k * 16 + j];
    tb[j] = fmaxf(a, 0.0f);
  }
  for (int j = 0; j < 16; ++j) {
    float a = reb2[j];
    for (int k = 0; k < 16; ++k) a += tb[k] * reW2[k * 16 + j];
    edge_attr[e * 16 + j] = attr[j] + a;
  }
  atomAddF(&cntf[s], 1.0f);
}

// h += acc/cnt ; acc = 0
__global__ void k_update(float* __restrict__ h0, float* __restrict__ h1,
                         float* __restrict__ acc0, float* __restrict__ acc1,
                         const float* __restrict__ cntf) {
  int i = blockIdx.x * blockDim.x + threadIdx.x;
  if (i >= NNODES * 28) return;
  int n = i / 28, c = i % 28;
  float cnt = fmaxf(cntf[n], 1.0f);
  if (c < 16) {
    h0[n * 16 + c] += acc0[n * 16 + c] / cnt;
    acc0[n * 16 + c] = 0.0f;
  } else {
    int cc = c - 16;
    h1[n * 12 + cc] += acc1[n * 12 + cc] / cnt;
    acc1[n * 12 + cc] = 0.0f;
  }
}

// final o3.Linear + sorter:  out[n, 4c+k] = k==0 ? f0[c] : f1[c][k-1]
__global__ void k_output(const float* __restrict__ h0, const float* __restrict__ h1,
                         const float* __restrict__ lin0, const float* __restrict__ lin1,
                         float* __restrict__ out) {
  int i = blockIdx.x * blockDim.x + threadIdx.x;
  if (i >= NNODES * 32) return;
  int n = i >> 5, c = i & 31;
  float f0 = 0.0f;
#pragma unroll
  for (int k = 0; k < 16; ++k) f0 += h0[n * 16 + k] * lin0[k * 32 + c];
  f0 *= 0.25f;  // 1/sqrt(16)
  float f1[3] = {0.0f, 0.0f, 0.0f};
#pragma unroll
  for (int v = 0; v < 4; ++v) {
    float w = lin1[v * 32 + c];
#pragma unroll
    for (int x = 0; x < 3; ++x) f1[x] += h1[n * 12 + v * 3 + x] * w;
  }
  float* o = out + (size_t)n * 128 + c * 4;
  o[0] = f0;
#pragma unroll
  for (int x = 0; x < 3; ++x) o[1 + x] = f1[x] * 0.5f;  // 1/sqrt(4)
}

// ---------------------------------------------------------------- main WMMA kernel

struct TileLDS {
  _Float16 A1[16 * 68];   // [edge_attr | h0[src] | h0[dst] | 0-pad]
  _Float16 A2[16 * 68];   // relu(GEMM1) activations
  float X0[16 * 17];      // h0[dst]
  float X1[16 * 13];      // h1[dst] [v*3+x]
  float Y1v[16 * 4];      // y1
  float DOTv[16 * 4];     // dot(x1,y1)/sqrt3
  float CRv[16 * 13];     // cross(x1,y1)/sqrt2
  int SRCv[16];
};

#if HAVE_TDM
// Issue one TDM descriptor: 1D copy of 7680 x 8B from global -> LDS.
__device__ inline void tdm_issue(const _Float16* src, _Float16* dst_lds) {
  unsigned long long ga = (unsigned long long)(uintptr_t)src;
  unsigned int la = (unsigned int)(uintptr_t)dst_lds;  // low 32 bits = LDS offset
  u32x4 g0;
  g0[0] = 1u;                                   // count=1, user descriptor
  g0[1] = la;                                   // lds_addr
  g0[2] = (unsigned int)ga;                     // global_addr[31:0]
  g0[3] = (unsigned int)((ga >> 32) & 0x01FFFFFFu) | (2u << 30);  // [56:32] + type=2
  i32x8 g1;
  g1[0] = (int)(3u << 16);                      // data_size=3 (8 bytes)
  g1[1] = (int)((WFRAG_QWORDS & 0xFFFFu) << 16);// tensor_dim0 lo16 (bits 63:48)
  g1[2] = (int)((WFRAG_QWORDS >> 16) | (1u << 16)); // tensor_dim0 hi16 | tensor_dim1=1
  g1[3] = (int)((unsigned)WFRAG_QWORDS << 16);  // tile_dim0 (bits 127:112)
  g1[4] = 0;                                    // tile_dim1=0, tile_dim2=0 (1D)
  g1[5] = WFRAG_QWORDS;                         // tensor_dim0_stride lo32
  g1[6] = 0;
  g1[7] = 0;
  i32x4 z4 = (i32x4)0;
#if defined(__clang_major__) && __clang_major__ >= 23
  i32x8 z8 = (i32x8)0;
  __builtin_amdgcn_tensor_load_to_lds(g0, g1, z4, z4, z8, 0);
#else
  __builtin_amdgcn_tensor_load_to_lds(g0, g1, z4, z4, 0);
#endif
}
#endif

__global__ __launch_bounds__(256) void k_layer(
    int layer,
    const float* __restrict__ h0, const float* __restrict__ h1,
    const float* __restrict__ edge_attr, const float* __restrict__ y1g,
    const int* __restrict__ ei,
    const _Float16* __restrict__ wfrag,
    const float* __restrict__ fc1b, const float* __restrict__ fc2b,
    float* __restrict__ acc0, float* __restrict__ acc1) {
  __shared__ TileLDS tiles[8];
  __shared__ _Float16 WLDS[WFRAG_PER_LAYER];   // this layer's weight fragments
  const int lane = threadIdx.x & 31;
  const int wave = threadIdx.x >> 5;
  TileLDS& T = tiles[wave];
  const int tile = blockIdx.x * 8 + wave;   // 8000 tiles of 16 edges, exact
  const int base = tile * 16;
  const int m = lane & 15;
  const int nl = m;
  const int hi = lane >> 4;
  const int e = base + m;
  const int srcn = ei[e];
  const int dstn = ei[NEDGES + e];
  const _Float16* wsrc = wfrag + (size_t)layer * WFRAG_PER_LAYER;

  // ---- kick off the weight DMA (or cooperative copy), overlap with gather
#if HAVE_TDM
  if (threadIdx.x < 32) tdm_issue(wsrc, WLDS);
#else
  {
    const uint4* s4 = (const uint4*)wsrc;
    uint4* d4 = (uint4*)WLDS;
    for (int i = threadIdx.x; i < WFRAG_QWORDS / 2; i += 256) d4[i] = s4[i];
  }
#endif

  // ---- stage gather into LDS (lanes 0-15: dst side; lanes 16-31: attr+src)
  if (hi == 0) {
    T.SRCv[m] = srcn;
    const float4* h0d = (const float4*)(h0 + dstn * 16);
#pragma unroll
    for (int q = 0; q < 4; ++q) {
      float4 v = h0d[q];
      T.X0[m * 17 + 4 * q + 0] = v.x; T.A1[m * 68 + 32 + 4 * q + 0] = (_Float16)v.x;
      T.X0[m * 17 + 4 * q + 1] = v.y; T.A1[m * 68 + 32 + 4 * q + 1] = (_Float16)v.y;
      T.X0[m * 17 + 4 * q + 2] = v.z; T.A1[m * 68 + 32 + 4 * q + 2] = (_Float16)v.z;
      T.X0[m * 17 + 4 * q + 3] = v.w; T.A1[m * 68 + 32 + 4 * q + 3] = (_Float16)v.w;
    }
    const float4* h1d = (const float4*)(h1 + dstn * 12);
    float x1v[12];
#pragma unroll
    for (int q = 0; q < 3; ++q) {
      float4 v = h1d[q];
      x1v[4 * q + 0] = v.x; x1v[4 * q + 1] = v.y;
      x1v[4 * q + 2] = v.z; x1v[4 * q + 3] = v.w;
    }
#pragma unroll
    for (int k = 0; k < 12; ++k) T.X1[m * 13 + k] = x1v[k];
    float yv[3];
#pragma unroll
    for (int x = 0; x < 3; ++x) { yv[x] = y1g[e * 3 + x]; T.Y1v[m * 4 + x] = yv[x]; }
#pragma unroll
    for (int v = 0; v < 4; ++v) {
      float ax = x1v[v * 3 + 0], ay = x1v[v * 3 + 1], az = x1v[v * 3 + 2];
      T.DOTv[m * 4 + v] = (ax * yv[0] + ay * yv[1] + az * yv[2]) * INV_S3;
      T.CRv[m * 13 + v * 3 + 0] = (ay * yv[2] - az * yv[1]) * INV_S2;
      T.CRv[m * 13 + v * 3 + 1] = (az * yv[0] - ax * yv[2]) * INV_S2;
      T.CRv[m * 13 + v * 3 + 2] = (ax * yv[1] - ay * yv[0]) * INV_S2;
    }
  } else {
    const float4* ea = (const float4*)(edge_attr + e * 16);
    const float4* h0s = (const float4*)(h0 + srcn * 16);
#pragma unroll
    for (int q = 0; q < 4; ++q) {
      float4 v = ea[q];
      T.A1[m * 68 + 4 * q + 0] = (_Float16)v.x;
      T.A1[m * 68 + 4 * q + 1] = (_Float16)v.y;
      T.A1[m * 68 + 4 * q + 2] = (_Float16)v.z;
      T.A1[m * 68 + 4 * q + 3] = (_Float16)v.w;
    }
#pragma unroll
    for (int q = 0; q < 4; ++q) {
      float4 v = h0s[q];
      T.A1[m * 68 + 16 + 4 * q + 0] = (_Float16)v.x;
      T.A1[m * 68 + 16 + 4 * q + 1] = (_Float16)v.y;
      T.A1[m * 68 + 16 + 4 * q + 2] = (_Float16)v.z;
      T.A1[m * 68 + 16 + 4 * q + 3] = (_Float16)v.w;
    }
#pragma unroll
    for (int k = 48; k < 64; ++k) T.A1[m * 68 + k] = (_Float16)0.0f;
  }

#if HAVE_TDM
  if (threadIdx.x < 32) __builtin_amdgcn_s_wait_tensorcnt(0);
#endif
  __syncthreads();

  const _Float16* f1b = WLDS;          // fc1 fragments: (s*4+jt)*512
  const _Float16* f2b = WLDS + 4096;   // fc2 fragments: (s*26+jt)*512

  // ---- GEMM1: [16,64] = relu( A1[16,64] x fc1p[64,64] + b1 )
  {
    v16h aS0 = load_afrag(T.A1, lane, 0);
    v16h aS1 = load_afrag(T.A1, lane, 1);
#pragma unroll
    for (int jt = 0; jt < 4; ++jt) {
      v8f c = {};
      c = wmma16(aS0, load_bfrag(f1b + (size_t)(0 * 4 + jt) * 512, lane), c);
      c = wmma16(aS1, load_bfrag(f1b + (size_t)(1 * 4 + jt) * 512, lane), c);
      float bias = fc1b[layer * 64 + jt * 16 + nl];
#pragma unroll
      for (int r = 0; r < 8; ++r) {
        float v = c[r] + bias;
        T.A2[(r + 8 * hi) * 68 + jt * 16 + nl] = (_Float16)fmaxf(v, 0.0f);
      }
    }
  }
  __syncthreads();

  // per-lane copies of y1 for its 8 output rows
  float y1r[8][3];
#pragma unroll
  for (int r = 0; r < 8; ++r)
#pragma unroll
    for (int x = 0; x < 3; ++x) y1r[r][x] = T.Y1v[(r + 8 * hi) * 4 + x];

  // ---- GEMM2 (26 N-tiles of the 416 TP weights) fused with TP application
  v16h a2S0 = load_afrag(T.A2, lane, 0);
  v16h a2S1 = load_afrag(T.A2, lane, 1);
  float o0r[8] = {};
  float o1r[8][3] = {};
#pragma unroll
  for (int jt = 0; jt < 26; ++jt) {
    v8f c = {};
    c = wmma16(a2S0, load_bfrag(f2b + (size_t)(0 * 26 + jt) * 512, lane), c);
    c = wmma16(a2S1, load_bfrag(f2b + (size_t)(1 * 26 + jt) * 512, lane), c);
    float bias = fc2b[layer * 416 + jt * 16 + nl];
    float w[8];
#pragma unroll
    for (int r = 0; r < 8; ++r) w[r] = c[r] + bias;

    if (jt < 16) {
      // w00[n_in=jt][n_out=nl] : o0 += x0[n_in]*w
#pragma unroll
      for (int r = 0; r < 8; ++r) o0r[r] += T.X0[(r + 8 * hi) * 17 + jt] * w[r];
    } else if (jt < 20) {
      // w011[n][o], col = n*4+o : n = 4*(jt-16)+(nl>>2), o = nl&3
      int n = 4 * (jt - 16) + (nl >> 2);
      float p[8];
#pragma unroll
      for (int r = 0; r < 8; ++r) p[r] = T.X0[(r + 8 * hi) * 17 + n] * w[r];
#pragma unroll
      for (int r = 0; r < 8; ++r) {
        p[r] += __shfl_xor(p[r], 4, 32);
        p[r] += __shfl_xor(p[r], 8, 32);
      }
      if (nl < 4) {
#pragma unroll
        for (int r = 0; r < 8; ++r)
#pragma unroll
          for (int x = 0; x < 3; ++x) o1r[r][x] += p[r] * y1r[r][x];
      }
    } else if (jt == 20 || jt == 25) {
      // w101 / w111 : col = n*4+o
      const float* SRCM = (jt == 20) ? T.X1 : T.CRv;
      int n = nl >> 2;
      float p[8][3];
#pragma unroll
      for (int r = 0; r < 8; ++r)
#pragma unroll
        for (int x = 0; x < 3; ++x)
          p[r][x] = SRCM[(r + 8 * hi) * 13 + n * 3 + x] * w[r];
#pragma unroll
      for (int r = 0; r < 8; ++r)
#pragma unroll
        for (int x = 0; x < 3; ++x) {
          p[r][x] += __shfl_xor(p[r][x], 4, 32);
          p[r][x] += __shfl_xor(p[r][x], 8, 32);
        }
      if (nl < 4) {
#pragma unroll
        for (int r = 0; r < 8; ++r)
#pragma unroll
          for (int x = 0; x < 3; ++x) o1r[r][x] += p[r][x];
      }
    } else {
      // w110[v][n_out], v = jt-21 : o0 += dot[v]*w
      int v = jt - 21;
#pragma unroll
      for (int r = 0; r < 8; ++r) o0r[r] += T.DOTv[(r + 8 * hi) * 4 + v] * w[r];
    }
  }

  // ---- scatter-add into per-node accumulators
#pragma unroll
  for (int r = 0; r < 8; ++r) {
    int sn = T.SRCv[r + 8 * hi];
    atomAddF(&acc0[sn * 16 + nl], A0SCALE * o0r[r]);
  }
  if (nl < 4) {
#pragma unroll
    for (int r = 0; r < 8; ++r) {
      int sn = T.SRCv[r + 8 * hi];
#pragma unroll
      for (int x = 0; x < 3; ++x)
        atomAddF(&acc1[sn * 12 + nl * 3 + x], A1SCALE * o1r[r][x]);
    }
  }
}

// ---------------------------------------------------------------- launcher

extern "C" void kernel_launch(void* const* d_in, const int* in_sizes, int n_in,
                              void* d_out, int out_size, void* d_ws, size_t ws_size,
                              hipStream_t stream) {
  const float* pos     = (const float*)d_in[0];
  const float* nfeats  = (const float*)d_in[1];
  const float* eain    = (const float*)d_in[2];
  const int*   ei      = (const int*)d_in[3];
  const float* encW    = (const float*)d_in[4];
  const float* encB    = (const float*)d_in[5];
  const float* neW1    = (const float*)d_in[6];
  const float* neB1    = (const float*)d_in[7];
  const float* neW2    = (const float*)d_in[8];
  const float* neB2    = (const float*)d_in[9];
  const float* eeW1    = (const float*)d_in[10];
  const float* eeB1    = (const float*)d_in[11];
  const float* eeW2    = (const float*)d_in[12];
  const float* eeB2    = (const float*)d_in[13];
  const float* reW1    = (const float*)d_in[14];
  const float* reB1    = (const float*)d_in[15];
  const float* reW2    = (const float*)d_in[16];
  const float* reB2    = (const float*)d_in[17];
  const float* fc1W    = (const float*)d_in[18];
  const float* fc1b    = (const float*)d_in[19];
  const float* fc2W    = (const float*)d_in[20];
  const float* fc2b    = (const float*)d_in[21];
  const float* lin0W   = (const float*)d_in[22];
  const float* lin1W   = (const float*)d_in[23];

  float* ws   = (float*)d_ws;
  float* h0   = ws;                       // N*16
  float* h1   = h0 + NNODES * 16;         // N*12
  float* acc0 = h1 + NNODES * 12;         // N*16
  float* acc1 = acc0 + NNODES * 16;       // N*12
  float* cntf = acc1 + NNODES * 12;       // N
  float* eatt = cntf + NNODES;            // E*16
  float* y1g  = eatt + NEDGES * 16;       // E*3
  _Float16* wfrag = (_Float16*)(y1g + NEDGES * 3);  // 2*30720 halves (aligned)

  // zero h1 | acc0 | acc1 | cntf (contiguous region)
  int nz = NNODES * (12 + 16 + 12 + 1);
  k_zero<<<(nz + 255) / 256, 256, 0, stream>>>(h1, nz);

  k_prep<<<(2 * WFRAG_PER_LAYER + 255) / 256, 256, 0, stream>>>(fc1W, fc2W, wfrag);

  k_node_embed<<<(NNODES + 127) / 128, 128, 0, stream>>>(
      nfeats, encW, encB, neW1, neB1, neW2, neB2, h0);

  k_edge_prep<<<(NEDGES + 127) / 128, 128, 0, stream>>>(
      pos, eain, ei, eeW1, eeB1, eeW2, eeB2, reW1, reB1, reW2, reB2, eatt, y1g, cntf);

  for (int l = 0; l < 2; ++l) {
    k_layer<<<NEDGES / (16 * 8), 256, 0, stream>>>(
        l, h0, h1, eatt, y1g, ei, wfrag, fc1b, fc2b, acc0, acc1);
    k_update<<<(NNODES * 28 + 255) / 256, 256, 0, stream>>>(h0, h1, acc0, acc1, cntf);
  }

  k_output<<<(NNODES * 32 + 255) / 256, 256, 0, stream>>>(h0, h1, lin0W, lin1W,
                                                          (float*)d_out);
}
